// GCN_53549652247110
// MI455X (gfx1250) — matrix operand order repaired and checked by reading, not compiled, not run on hardware
//
#include <hip/hip_runtime.h>
#include <hip/hip_bf16.h>
#include <math.h>

// ---------------------------------------------------------------------------
// MI455X / gfx1250 GCN forward. bf16 split-precision WMMA GEMMs
// (v_wmma_f32_16x16x32_bf16, 3 wmma per 32-wide k-step for ~fp32 accuracy),
// streaming edge scatter with global f32 atomics, fused LN+GELU epilogues.
// ---------------------------------------------------------------------------

typedef __attribute__((ext_vector_type(16))) __bf16 v16bf;
typedef __attribute__((ext_vector_type(8)))  __bf16 v8bf;
typedef __attribute__((ext_vector_type(8)))  float  v8f;

__device__ __forceinline__ float gelu_f(float x) {
  return 0.5f * x * (1.0f + erff(x * 0.70710678118654752f));
}

// ---- weight pre-pack into WMMA B-fragment order --------------------------
// packed[((kp*NT + nt)*32 + lane)*16 + e]  holds  W[kp*32 + (lane>>4)*16 + e][nt*16 + (lane&15)]
__global__ __launch_bounds__(256) void pack_weights(
    const float* __restrict__ W, __bf16* __restrict__ Phi, __bf16* __restrict__ Plo,
    int K, int Nc) {
  int idx = blockIdx.x * 256 + threadIdx.x;
  int total = K * Nc;
  if (idx >= total) return;
  int e    = idx & 15;
  int lane = (idx >> 4) & 31;
  int tile = idx >> 9;
  int NT   = Nc >> 4;
  int nt   = tile % NT;
  int kp   = tile / NT;
  int k = kp * 32 + (lane >> 4) * 16 + e;
  int n = nt * 16 + (lane & 15);
  float w = W[(size_t)k * Nc + n];
  __bf16 h = (__bf16)w;
  Phi[idx] = h;
  Plo[idx] = (__bf16)(w - (float)h);
}

// ---- fp32 -> (hi,lo) bf16 planes -----------------------------------------
__global__ __launch_bounds__(256) void split_bf16(
    const float* __restrict__ X, __bf16* __restrict__ hi, __bf16* __restrict__ lo, size_t n) {
  size_t i = (size_t)blockIdx.x * 256 + threadIdx.x;
  if (i >= n) return;
  float v = X[i];
  __bf16 h = (__bf16)v;
  hi[i] = h;
  lo[i] = (__bf16)(v - (float)h);
}

__global__ __launch_bounds__(256) void add_split(
    const float* __restrict__ A, const float* __restrict__ B,
    __bf16* __restrict__ hi, __bf16* __restrict__ lo, size_t n) {
  size_t i = (size_t)blockIdx.x * 256 + threadIdx.x;
  if (i >= n) return;
  float v = A[i] + B[i];
  __bf16 h = (__bf16)v;
  hi[i] = h;
  lo[i] = (__bf16)(v - (float)h);
}

__global__ __launch_bounds__(256) void zero_f32(float* __restrict__ p, int n) {
  int i = blockIdx.x * 256 + threadIdx.x;
  if (i < n) p[i] = 0.0f;
}

__global__ __launch_bounds__(256) void degree_accum(
    const int* __restrict__ dst, float* __restrict__ deg, int E) {
  int e = blockIdx.x * 256 + threadIdx.x;
  if (e < E) atomicAdd(&deg[dst[e]], 1.0f);
}

__global__ __launch_bounds__(256) void make_dinv(
    const float* __restrict__ deg, float* __restrict__ dinv, int n) {
  int i = blockIdx.x * 256 + threadIdx.x;
  if (i < n) dinv[i] = rsqrtf(deg[i] + 1.0f);  // +1 self loop
}

// ---- WMMA GEMM: C[M x Nc] = A[M x K] * B, A given as bf16 hi/lo planes ----
// Each wave: one 16-row M tile x four 16-col N tiles. flags: 1=bias, 2=gelu.
__device__ __forceinline__ v16bf load_a_frag(
    const __bf16* __restrict__ A, size_t rowoff, int kb, int half) {
  v8bf c0 = *(const v8bf*)(A + rowoff + kb + half * 16);       // K = kb + half*16 .. +7
  v8bf c1 = *(const v8bf*)(A + rowoff + kb + half * 16 + 8);   // K = kb+half*16+8 .. +15
  v16bf f;
#pragma unroll
  for (int i = 0; i < 8; ++i) { f[i] = c0[i]; f[i + 8] = c1[i]; }
  return f;
}

__global__ __launch_bounds__(256, 2) void gemm_bf16s(
    const __bf16* __restrict__ Ahi, const __bf16* __restrict__ Alo,
    const __bf16* __restrict__ Bhi, const __bf16* __restrict__ Blo,
    float* __restrict__ C, const float* __restrict__ bias,
    int M, int K, int Nc, int flags) {
  const int lane = threadIdx.x & 31;
  const int wave = threadIdx.x >> 5;
  const int mt   = blockIdx.x * 8 + wave;          // 16-row tile index
  if (mt * 16 >= M) return;
  const int half = lane >> 4;
  const int r    = lane & 15;
  const int NT   = Nc >> 4;
  const int nt0  = blockIdx.y * 4;                 // first of 4 n-tiles
  const size_t rowoff = (size_t)(mt * 16 + r) * K;

  v8f acc[4] = {};
  for (int kb = 0; kb < K; kb += 32) {
    v16bf ah = load_a_frag(Ahi, rowoff, kb, half);
    v16bf al = load_a_frag(Alo, rowoff, kb, half);
    const int kp = kb >> 5;
    const size_t tb = (((size_t)kp * NT + nt0) * 32 + lane) * 16;
#pragma unroll
    for (int t = 0; t < 4; ++t) {
      v16bf bh = *(const v16bf*)(Bhi + tb + (size_t)t * 512);
      v16bf bl = *(const v16bf*)(Blo + tb + (size_t)t * 512);
      acc[t] = __builtin_amdgcn_wmma_f32_16x16x32_bf16(false, ah, false, bh,
                                                       (short)0, acc[t], false, false);
      acc[t] = __builtin_amdgcn_wmma_f32_16x16x32_bf16(false, al, false, bh,
                                                       (short)0, acc[t], false, false);
      acc[t] = __builtin_amdgcn_wmma_f32_16x16x32_bf16(false, ah, false, bl,
                                                       (short)0, acc[t], false, false);
    }
  }
#pragma unroll
  for (int t = 0; t < 4; ++t) {
    int n = (nt0 + t) * 16 + r;
    float b = (flags & 1) ? bias[n] : 0.0f;
#pragma unroll
    for (int i = 0; i < 8; ++i) {                   // VGPR i -> row mt*16 + half*8 + i
      float v = acc[t][i] + b;
      if (flags & 2) v = gelu_f(v);
      C[(size_t)(mt * 16 + half * 8 + i) * Nc + n] = v;
    }
  }
}

// ---- GCN conv pieces (d_out fixed at 256) ---------------------------------
__global__ __launch_bounds__(256) void conv_init(
    const float* __restrict__ HW, const float* __restrict__ SP,
    const float* __restrict__ dinv, const float* __restrict__ bc,
    const float* __restrict__ bp, float* __restrict__ OUT, int N) {
  size_t idx = (size_t)blockIdx.x * 256 + threadIdx.x;
  if (idx >= (size_t)N * 256) return;
  int row = (int)(idx >> 8);
  int c   = (int)(idx & 255);
  float di = dinv[row];
  OUT[idx] = di * di * HW[idx] + SP[idx] + bc[c] + bp[c];
}

__global__ __launch_bounds__(256) void scatter_edges(
    const float* __restrict__ HW, float* __restrict__ OUT,
    const int* __restrict__ src, const int* __restrict__ dst,
    const float* __restrict__ dinv, int E) {
  const int c = threadIdx.x;
  const int base = blockIdx.x * 4;
  for (int j = 0; j < 4; ++j) {
    int e = base + j;
    if (e >= E) return;
    if (e + 1 < E)  // hint: prefetch next source row (global_prefetch_b8)
      __builtin_prefetch(&HW[(size_t)src[e + 1] * 256 + c], 0, 1);
    int s = src[e], t = dst[e];
    float nm = dinv[s] * dinv[t];
    atomicAdd(&OUT[(size_t)t * 256 + c], HW[(size_t)s * 256 + c] * nm);
  }
}

// ---- fused LayerNorm + GELU + bf16 splits of both h and skip --------------
__device__ __forceinline__ float block_sum256(float v) {
  __shared__ float red[8];
  int lane = threadIdx.x & 31, w = threadIdx.x >> 5;
#pragma unroll
  for (int o = 16; o; o >>= 1) v += __shfl_down(v, o, 32);
  __syncthreads();
  if (lane == 0) red[w] = v;
  __syncthreads();
  float s = red[0];
#pragma unroll
  for (int i = 1; i < 8; ++i) s += red[i];
  return s;
}

__global__ __launch_bounds__(256) void ln_gelu_split(
    const float* __restrict__ X, const float* __restrict__ g, const float* __restrict__ be,
    float* __restrict__ Hf, __bf16* __restrict__ Hhi, __bf16* __restrict__ Hlo,
    __bf16* __restrict__ Shi, __bf16* __restrict__ Slo) {
  const int c = threadIdx.x;
  const size_t idx = (size_t)blockIdx.x * 256 + c;
  float v = X[idx];
  __bf16 sh = (__bf16)v;                 // skip = pre-LN value
  Shi[idx] = sh;
  Slo[idx] = (__bf16)(v - (float)sh);
  float m = block_sum256(v) * (1.0f / 256.0f);
  float d = v - m;
  float var = block_sum256(d * d) * (1.0f / 256.0f);
  float y = d * rsqrtf(var + 1e-5f) * g[c] + be[c];
  y = gelu_f(y);
  Hf[idx] = y;
  __bf16 hh = (__bf16)y;
  Hhi[idx] = hh;
  Hlo[idx] = (__bf16)(y - (float)hh);
}

// ---- row-wise log_softmax over 64 classes, in place -----------------------
__global__ __launch_bounds__(64) void log_softmax64(float* __restrict__ Z) {
  __shared__ float sm[64];
  const int c = threadIdx.x;
  const size_t idx = (size_t)blockIdx.x * 64 + c;
  float z = Z[idx];
  sm[c] = z; __syncthreads();
  for (int s = 32; s; s >>= 1) { if (c < s) sm[c] = fmaxf(sm[c], sm[c + s]); __syncthreads(); }
  float mx = sm[0]; __syncthreads();
  sm[c] = expf(z - mx); __syncthreads();
  for (int s = 32; s; s >>= 1) { if (c < s) sm[c] += sm[c + s]; __syncthreads(); }
  float ls = logf(sm[0]);
  Z[idx] = z - mx - ls;
}

// ---------------------------------------------------------------------------
extern "C" void kernel_launch(void* const* d_in, const int* in_sizes, int n_in,
                              void* d_out, int out_size, void* d_ws, size_t ws_size,
                              hipStream_t stream) {
  if (n_in < 26) return;
  const float* x  = (const float*)d_in[0];
  const int*   ei = (const int*)d_in[1];
  const int N = in_sizes[0] / 128;
  const int E = in_sizes[1] / 2;
  const int* src = ei;
  const int* dst = ei + E;
  const float* Wc[3] = {(const float*)d_in[2],  (const float*)d_in[8],  (const float*)d_in[14]};
  const float* bc[3] = {(const float*)d_in[3],  (const float*)d_in[9],  (const float*)d_in[15]};
  const float* Wp[3] = {(const float*)d_in[4],  (const float*)d_in[10], (const float*)d_in[16]};
  const float* bp[3] = {(const float*)d_in[5],  (const float*)d_in[11], (const float*)d_in[17]};
  const float* gg[3] = {(const float*)d_in[6],  (const float*)d_in[12], (const float*)d_in[18]};
  const float* be[3] = {(const float*)d_in[7],  (const float*)d_in[13], (const float*)d_in[19]};
  const float* W_in = (const float*)d_in[20];
  const float* b_in = (const float*)d_in[21];
  const float* Wf1  = (const float*)d_in[22];
  const float* bf1  = (const float*)d_in[23];
  const float* Wf2  = (const float*)d_in[24];
  const float* bf2  = (const float*)d_in[25];
  float* out = (float*)d_out;

  // ---- workspace carve-up (~440 MB total) ----
  char* ws = (char*)d_ws;
  size_t off = 0;
  auto alloc = [&](size_t bytes) -> void* {
    off = (off + 255) & ~(size_t)255;
    void* p = ws + off;
    off += bytes;
    return p;
  };
  struct PW { __bf16 *hi, *lo; };
  auto allocPW = [&](int K, int Nc) {
    PW p;
    p.hi = (__bf16*)alloc((size_t)K * Nc * 2);
    p.lo = (__bf16*)alloc((size_t)K * Nc * 2);
    return p;
  };
  const int convK[3] = {128, 256, 256};
  PW pWc[3], pWp[3];
  for (int l = 0; l < 3; ++l) { pWc[l] = allocPW(convK[l], 256); pWp[l] = allocPW(convK[l], 256); }
  PW pWin = allocPW(128, 256);
  PW pWf1 = allocPW(256, 512);
  PW pWf2 = allocPW(512, 64);

  __bf16* xhi    = (__bf16*)alloc((size_t)N * 128 * 2);
  __bf16* xlo    = (__bf16*)alloc((size_t)N * 128 * 2);
  __bf16* curhi  = (__bf16*)alloc((size_t)N * 256 * 2);
  __bf16* curlo  = (__bf16*)alloc((size_t)N * 256 * 2);
  __bf16* skiphi = (__bf16*)alloc((size_t)N * 256 * 2);
  __bf16* skiplo = (__bf16*)alloc((size_t)N * 256 * 2);
  float*  HW     = (float*)alloc((size_t)N * 256 * 4);
  float*  SP     = (float*)alloc((size_t)N * 256 * 4);
  float*  OUTb   = (float*)alloc((size_t)N * 256 * 4);
  float*  Hbuf   = (float*)alloc((size_t)N * 256 * 4);
  float*  F      = (float*)alloc((size_t)N * 512 * 4);
  float*  deg    = (float*)alloc((size_t)N * 4);
  float*  dinv   = (float*)alloc((size_t)N * 4);
  // F splits reuse HW/OUTb (free by then): each needs N*512*2 = N*256*4 bytes
  __bf16* Fhi = (__bf16*)HW;
  __bf16* Flo = (__bf16*)OUTb;
  (void)ws_size; (void)out_size;

  auto pack = [&](const float* W, PW p, int K, int Nc) {
    int tot = K * Nc;
    pack_weights<<<(tot + 255) / 256, 256, 0, stream>>>(W, p.hi, p.lo, K, Nc);
  };
  for (int l = 0; l < 3; ++l) { pack(Wc[l], pWc[l], convK[l], 256); pack(Wp[l], pWp[l], convK[l], 256); }
  pack(W_in, pWin, 128, 256);
  pack(Wf1, pWf1, 256, 512);
  pack(Wf2, pWf2, 512, 64);

  const size_t nx = (size_t)N * 128;
  split_bf16<<<(int)((nx + 255) / 256), 256, 0, stream>>>(x, xhi, xlo, nx);

  zero_f32<<<(N + 255) / 256, 256, 0, stream>>>(deg, N);
  degree_accum<<<(E + 255) / 256, 256, 0, stream>>>(dst, deg, E);
  make_dinv<<<(N + 255) / 256, 256, 0, stream>>>(deg, dinv, N);

  auto gemm = [&](const __bf16* Ahi, const __bf16* Alo, PW B, float* C,
                  const float* bias, int M, int K, int Nc, int flags) {
    dim3 grid((M / 16 + 7) / 8, Nc / 64);
    gemm_bf16s<<<grid, 256, 0, stream>>>(Ahi, Alo, B.hi, B.lo, C, bias, M, K, Nc, flags);
  };

  const size_t nTot = (size_t)N * 256;
  const int ewBlocks = (int)((nTot + 255) / 256);

  const __bf16 *chi = xhi, *clo = xlo, *shi = xhi, *slo = xlo;
  for (int l = 0; l < 3; ++l) {
    gemm(chi, clo, pWc[l], HW, nullptr, N, convK[l], 256, 0);   // hw = h @ Wc
    gemm(shi, slo, pWp[l], SP, nullptr, N, convK[l], 256, 0);   // sp = skip @ Wp
    conv_init<<<ewBlocks, 256, 0, stream>>>(HW, SP, dinv, bc[l], bp[l], OUTb, N);
    scatter_edges<<<(E + 3) / 4, 256, 0, stream>>>(HW, OUTb, src, dst, dinv, E);
    ln_gelu_split<<<N, 256, 0, stream>>>(OUTb, gg[l], be[l], Hbuf, curhi, curlo, skiphi, skiplo);
    chi = curhi; clo = curlo; shi = skiphi; slo = skiplo;
  }

  // long residual: hf = x @ W_in + b_in + h3
  gemm(xhi, xlo, pWin, SP, b_in, N, 128, 256, 1);
  add_split<<<ewBlocks, 256, 0, stream>>>(SP, Hbuf, skiphi, skiplo, nTot);

  // f = gelu(hf @ Wf1 + bf1)
  gemm(skiphi, skiplo, pWf1, F, bf1, N, 256, 512, 3);
  const size_t nF = (size_t)N * 512;
  split_bf16<<<(int)((nF + 255) / 256), 256, 0, stream>>>(F, Fhi, Flo, nF);

  // z = f @ Wf2 + bf2 -> d_out, then log_softmax in place
  gemm(Fhi, Flo, pWf2, out, bf2, N, 512, 64, 1);
  log_softmax64<<<N, 64, 0, stream>>>(out);
}
